// LocallyConnectedLayer_584115552260
// MI455X (gfx1250) — matrix operand order repaired and checked by reading, not compiled, not run on hardware
//
#include <hip/hip_runtime.h>
#include <hip/hip_bf16.h>

typedef __attribute__((ext_vector_type(16))) __bf16 v16bf;
typedef __attribute__((ext_vector_type(8)))  __bf16 v8bf;
typedef __attribute__((ext_vector_type(2)))  __bf16 v2bf;
typedef __attribute__((ext_vector_type(8)))  float  v8f;

// Problem constants (from reference): B=128, C=64, H=W=32, O=64, OH=OW=30, K=3
#define BN   128
#define CN   64
#define ON   64
#define HN   32
#define WN   32
#define OHN  30
#define OWN  30
#define SPN  (OHN * OWN)        // 900 spatial sites
#define LDP  72                 // padded LDS row stride (bf16 elems), 144B = 16B-aligned

// 16x32 bf16 A-fragment row (per-lane), contiguous layout (matches B):
//   lanes 0-15 : elems[0..15]=K base+0..15
//   lanes16-31 : same with base += 16
__device__ __forceinline__ v16bf load_a_frag(const __bf16* row, int base) {
    v8bf lo = *(const v8bf*)(row + base);
    v8bf hi = *(const v8bf*)(row + base + 8);
    return __builtin_shufflevector(lo, hi, 0,1,2,3,4,5,6,7,8,9,10,11,12,13,14,15);
}

// 32x16 bf16 B-fragment column (per-lane), contiguous layout (matches A):
//   lanes 0-15 : K = base+0..15 ;  lanes 16-31 : K = base+16..31
__device__ __forceinline__ v16bf load_b_frag(const __bf16* row, int base) {
    v8bf lo = *(const v8bf*)(row + base);
    v8bf hi = *(const v8bf*)(row + base + 8);
    return __builtin_shufflevector(lo, hi, 0,1,2,3,4,5,6,7,8,9,10,11,12,13,14,15);
}

__global__ __launch_bounds__(256)
void lc_wmma_kernel(const float* __restrict__ x,
                    const float* __restrict__ w,
                    const float* __restrict__ bias,
                    float* __restrict__ out) {
    const int sp   = blockIdx.x;          // spatial site, 0..899
    const int oi   = sp / OWN;
    const int oj   = sp % OWN;
    const int t    = threadIdx.x;
    const int wave = t >> 5;              // 0..7 -> M-tile (batch rows wave*16..+15)
    const int lane = t & 31;
    const int l15  = lane & 15;
    const int lh   = lane >> 4;           // lane-half
    const int b0   = wave * 16;

    // A panels, one per kw: sA[kw][b][c] = bf16(x[b,c,i+kh,j+kw]).   3*128*72*2 = 54 KB
    __shared__ __align__(16) __bf16 sA[3][BN][LDP];
    // B^T panels, one per kw: sBt[kw][o][c] = bf16(weight[c,o,i,j,kh,kw]).  3*64*72*2 = 27 KB
    __shared__ __align__(16) __bf16 sBt[3][ON][LDP];

    const v8f vzero = {0.f, 0.f, 0.f, 0.f, 0.f, 0.f, 0.f, 0.f};
    v8f acc[4];
#pragma unroll
    for (int n = 0; n < 4; ++n) acc[n] = vzero;

    for (int kh = 0; kh < 3; ++kh) {
        const int hh = oi + kh;
        __syncthreads();   // previous kh's WMMAs done before overwriting panels

        // ---- stage x: 8192 (b,c) pairs, 2 adjacent-c pairs/thread/iter.
        // One b96 load per pair (covers kw=0..2); packed cvt + b32 LDS store.
#pragma unroll 2
        for (int e = 0; e < 16; ++e) {
            const int p0 = e * 512 + t * 2;         // even pair index: p = b*64 + c
            const float* xp0 = x + p0 * (HN * WN) + hh * WN + oj;
            const float* xp1 = xp0 + (HN * WN);
            const float a0 = xp0[0], a1 = xp0[1], a2 = xp0[2];
            const float b0f = xp1[0], b1f = xp1[1], b2f = xp1[2];
            const int bb = p0 >> 6;
            const int cc = p0 & 63;                 // even -> 4B-aligned b32 store
            v2bf pk;
            pk.x = (__bf16)a0;  pk.y = (__bf16)b0f; *(v2bf*)&sA[0][bb][cc] = pk;
            pk.x = (__bf16)a1;  pk.y = (__bf16)b1f; *(v2bf*)&sA[1][bb][cc] = pk;
            pk.x = (__bf16)a2;  pk.y = (__bf16)b2f; *(v2bf*)&sA[2][bb][cc] = pk;
        }

        // ---- stage weight: 4096 (c,o) pairs as 2048 adjacent-c pairs.
        // One b96 load per (c,o) covers kw=0..2 (contiguous kh*3+{0,1,2}).
#pragma unroll 2
        for (int e = 0; e < 8; ++e) {
            const int idx = e * 256 + t;            // enumerates (cpair, o)
            const int o  = idx & 63;
            const int c0 = (idx >> 6) * 2;          // even c
            const float* wp0 = w + ((c0 * 64 + o) * SPN + sp) * 9 + kh * 3;
            const float* wp1 = wp0 + 64 * SPN * 9;  // c0+1
            const float a0 = wp0[0], a1 = wp0[1], a2 = wp0[2];
            const float b0f = wp1[0], b1f = wp1[1], b2f = wp1[2];
            v2bf pk;
            pk.x = (__bf16)a0;  pk.y = (__bf16)b0f; *(v2bf*)&sBt[0][o][c0] = pk;
            pk.x = (__bf16)a1;  pk.y = (__bf16)b1f; *(v2bf*)&sBt[1][o][c0] = pk;
            pk.x = (__bf16)a2;  pk.y = (__bf16)b2f; *(v2bf*)&sBt[2][o][c0] = pk;
        }
        __syncthreads();   // staging visible before compute

        // ---- 3 kw panels; per panel preload ALL fragments (distinct regs) so the
        // 20 ds_load_b128 issue as a clause and the 8 WMMAs overlap the DS pipe
        // instead of each waiting on s_wait_dscnt 0.
#pragma unroll
        for (int kw = 0; kw < 3; ++kw) {
            const __bf16* arow = &sA[kw][b0 + l15][0];
            v16bf af[2];
            v16bf bf[2][4];
#pragma unroll
            for (int kc = 0; kc < 2; ++kc) {
                af[kc] = load_a_frag(arow, kc * 32 + lh * 16);
#pragma unroll
                for (int n = 0; n < 4; ++n)
                    bf[kc][n] = load_b_frag(&sBt[kw][n * 16 + l15][0],
                                            kc * 32 + lh * 16);
            }
#pragma unroll
            for (int kc = 0; kc < 2; ++kc) {
#pragma unroll
                for (int n = 0; n < 4; ++n) {
                    acc[n] = __builtin_amdgcn_wmma_f32_16x16x32_bf16(
                        false, af[kc], false, bf[kc][n], (short)0, acc[n],
                        false, false);
                }
            }
        }
    }

    // ---- epilogue: add bias[o, oi, oj], scatter to out[b, o, oi, oj]
#pragma unroll
    for (int n = 0; n < 4; ++n) {
        const int o  = n * 16 + l15;
        const float bv = bias[o * SPN + sp];
#pragma unroll
        for (int r = 0; r < 8; ++r) {
            const int bb = b0 + r + lh * 8;
            out[(bb * ON + o) * SPN + sp] = acc[n][r] + bv;
        }
    }
}

extern "C" void kernel_launch(void* const* d_in, const int* in_sizes, int n_in,
                              void* d_out, int out_size, void* d_ws, size_t ws_size,
                              hipStream_t stream) {
    const float* x    = (const float*)d_in[0];   // [128,64,32,32]
    const float* wgt  = (const float*)d_in[1];   // [64,64,30,30,3,3]
    const float* bias = (const float*)d_in[2];   // [64,30,30]
    float* out        = (float*)d_out;           // [128,64,30,30]
    (void)in_sizes; (void)n_in; (void)out_size; (void)d_ws; (void)ws_size;

    lc_wmma_kernel<<<SPN, 256, 0, stream>>>(x, wgt, bias, out);
}
